// M3GNet_84576495993542
// MI455X (gfx1250) — compile-verified, matrix operand order checked by reading, not run on hardware
//
#include <hip/hip_runtime.h>
#include <hip/hip_bf16.h>
#include <math.h>

// ---------------- types for WMMA ----------------
typedef __attribute__((ext_vector_type(16))) __bf16 v16bf;
typedef __attribute__((ext_vector_type(8)))  __bf16 v8bf;
typedef __attribute__((ext_vector_type(8)))  float  v8f;

#define FD   128
#define NRr  4
#define Ll   4
#define CUTf 5.0f
#define CUT3f 4.0f
#define PI_F 3.14159265358979323846f

// ---------------- small device helpers ----------------
__device__ __forceinline__ unsigned short f2bf(float f) {
    unsigned int u = __float_as_uint(f);
    unsigned int r = (u + 0x7FFFu + ((u >> 16) & 1u)) >> 16;   // RNE
    return (unsigned short)r;
}
__device__ __forceinline__ unsigned int pack2bf(float lo, float hi) {
    return (unsigned int)f2bf(lo) | ((unsigned int)f2bf(hi) << 16);
}
__device__ __forceinline__ float sigmoidf_(float x) { return 1.0f / (1.0f + __expf(-x)); }
__device__ __forceinline__ float swishf_(float x) { return x * sigmoidf_(x); }
__device__ __forceinline__ float cutoff_f(float r, float rc) {
    float x = r / rc;
    float x2 = x * x, x3 = x2 * x;
    return 1.0f - 6.0f * x3 * x2 + 15.0f * x2 * x2 - 10.0f * x3;
}

// A-fragment: 16xK row-major bf16 tile in LDS (ldk elements per row).
// Wave32 layout -> two contiguous 16-byte loads per lane:
//   p=0..7  : k = kbase + 8*half + 0..7
//   p=8..15 : k = kbase + 16 + 8*half + 0..7
__device__ __forceinline__ v16bf load_a_frag(const unsigned short* __restrict__ A,
                                             int ldk, int kbase, int lane) {
    int m = lane & 15, h = lane >> 4;
    const unsigned short* row = A + m * ldk + kbase + 8 * h;
    union { v16bf v; v8bf half[2]; } u;
    u.half[0] = *(const v8bf*)(row);
    u.half[1] = *(const v8bf*)(row + 16);
    return u.v;
}

// B-fragment from a pre-swizzled (fragment-major) weight tile:
// layout: [(t*nkt + kt)*32 + lane][16 contiguous bf16]
__device__ __forceinline__ v16bf load_b_packed(const unsigned short* __restrict__ Wp,
                                               int t, int kt, int nkt, int lane) {
    return *(const v16bf*)(Wp + ((((t * nkt + kt) * 32) + lane) << 4));
}

// ---------------- weight pre-swizzle (run once per launch) ----------------
// src: nmat stacked (Ks x 128) row-major f32 matrices.
// dst: nmat stacked packed tiles, K padded to nkt*32 with zeros.
__global__ void k_pack_wB(const float* __restrict__ src, unsigned short* __restrict__ dst,
                          int nmat, int Ks, int nkt) {
    int i = blockIdx.x * 256 + threadIdx.x;
    int per = nkt * 8 * 32;                 // fragment-lanes per matrix
    if (i >= nmat * per) return;
    int m = i / per, r = i % per;
    int lane = r & 31;
    int q = r >> 5;                         // t*nkt + kt
    int kt = q % nkt, t = q / nkt;
    const float* S = src + (size_t)m * Ks * FD;
    unsigned short* D = dst + (size_t)m * per * 16 + (size_t)r * 16;
    int n = t * 16 + (lane & 15);
    int h = lane >> 4;
    int kb = kt * 32;
#pragma unroll
    for (int p = 0; p < 16; ++p) {
        int vg = p >> 1, w = p & 1;
        int k = kb + ((vg & 3) * 2) + ((vg >> 2) * 16) + 8 * h + w;
        D[p] = (k < Ks) ? f2bf(S[k * FD + n]) : (unsigned short)0;
    }
}

// ---------------- one-time prep kernels ----------------
__global__ void k_offsets(const int* __restrict__ tnb, const int* __restrict__ tna,
                          int G, int* __restrict__ boff, int* __restrict__ acum) {
    if (blockIdx.x == 0 && threadIdx.x == 0) {
        int cb = 0, ca = 0;
        for (int g = 0; g < G; ++g) {
            boff[g] = cb; cb += tnb[g];
            ca += tna[g]; acum[g] = ca;     // inclusive
        }
    }
}

__global__ void k_tbi(const int* __restrict__ tbi, const int* __restrict__ boff,
                      const int* __restrict__ acum, int G, int A,
                      int* __restrict__ idx_ij, int* __restrict__ idx_ik) {
    int a = blockIdx.x * 256 + threadIdx.x;
    if (a >= A) return;
    int lo = 0, hi = G - 1;
    while (lo < hi) { int mid = (lo + hi) >> 1; if (acum[mid] > a) hi = mid; else lo = mid + 1; }
    int off = boff[lo];
    idx_ij[a] = tbi[a * 2 + 0] + off;
    idx_ik[a] = tbi[a * 2 + 1] + off;
}

__global__ void k_rbf(const float* __restrict__ dist, float* __restrict__ ef0, int E) {
    int e = blockIdx.x * 256 + threadIdx.x;
    if (e >= E) return;
    float r = dist[e];
    float c = sqrtf(2.0f / CUTf) / r;
#pragma unroll
    for (int n = 1; n <= NRr + 1; ++n)
        ef0[e * 5 + (n - 1)] = c * __sinf((float)n * PI_F * r / CUTf);
}

__global__ void k_edge_enc(const float* __restrict__ ef0, const float* __restrict__ Wenc,
                           const float* __restrict__ benc, float* __restrict__ e, int E) {
    int i = blockIdx.x * 256 + threadIdx.x;
    if (i >= E * FD) return;
    int ed = i >> 7, f = i & 127;
    float s = benc[f];
#pragma unroll
    for (int q = 0; q < 5; ++q) s += ef0[ed * 5 + q] * Wenc[q * FD + f];
    e[i] = swishf_(s);
}

__global__ void k_vinit(const int* __restrict__ z, const float* __restrict__ emb,
                        float* __restrict__ v, int N) {
    int i = blockIdx.x * 256 + threadIdx.x;
    if (i >= N * FD) return;
    v[i] = emb[z[i >> 7] * FD + (i & 127)];
}

__global__ void k_angle(const float* __restrict__ norm_ik, const float* __restrict__ cosang,
                        float* __restrict__ ang, float* __restrict__ fc_ik, int A) {
    int a = blockIdx.x * 256 + threadIdx.x;
    if (a >= A) return;
    float r = norm_ik[a], ct = cosang[a];
    float fc = cutoff_f(r, CUT3f);
    float P[4];
    P[0] = 1.0f; P[1] = ct;
    P[2] = 0.5f * (3.0f * ct * ct - 1.0f);
    P[3] = 0.5f * (5.0f * ct * ct * ct - 3.0f * ct);
    fc_ik[a] = fc;
#pragma unroll
    for (int n = 1; n <= NRr; ++n) {
        float rad = __sinf((float)n * PI_F * r / CUT3f) / r;
#pragma unroll
        for (int l = 0; l < Ll; ++l)
            ang[a * 16 + (n - 1) * Ll + l] = rad * P[l] * fc;
    }
}

// ---------------- gate + msg + scatter (WMMA) ----------------
__global__ __launch_bounds__(128) void k_gate_msg(
    const float* __restrict__ v, const int* __restrict__ recv,
    const int* __restrict__ idx_ik, const int* __restrict__ idx_ij,
    const unsigned short* __restrict__ WgP, const float* __restrict__ bg,
    const float* __restrict__ ang, const unsigned short* __restrict__ WangP,
    const float* __restrict__ fc_ik, float* __restrict__ agg, int A) {
    __shared__ __align__(32) unsigned short Wlds[FD * FD];     // 32 KB packed Wg
    __shared__ __align__(32) unsigned short WaLds[8 * 32 * 16]; //  8 KB packed Wangle
    __shared__ __align__(16) unsigned short Alds[64 * FD];     // 16 KB v rows
    __shared__ __align__(16) unsigned short AngLds[64 * 32];   //  4 KB ang rows (K padded to 32)
    __shared__ int krow[64];
    __shared__ int jrow[64];

    int tid = threadIdx.x;
    int base = blockIdx.x * 64;

    for (int i = tid; i < 2048; i += 128) ((uint4*)Wlds)[i] = ((const uint4*)WgP)[i];
    for (int i = tid; i < 512;  i += 128) ((uint4*)WaLds)[i] = ((const uint4*)WangP)[i];
    if (tid < 64) {
        int row = base + tid; if (row >= A) row = A - 1;
        krow[tid] = recv[idx_ik[row]];
        jrow[tid] = idx_ij[row];
    }
    __syncthreads();
    // stage v[k_atom] rows -> bf16 (8 floats -> one uint4 b128 store)
    for (int i = tid; i < 64 * 16; i += 128) {
        int r = i >> 4, g = i & 15;
        const float4* s4 = (const float4*)(v + (size_t)krow[r] * FD + g * 8);
        float4 x0 = s4[0], x1 = s4[1];
        uint4 u;
        u.x = pack2bf(x0.x, x0.y); u.y = pack2bf(x0.z, x0.w);
        u.z = pack2bf(x1.x, x1.y); u.w = pack2bf(x1.z, x1.w);
        *(uint4*)(Alds + r * FD + g * 8) = u;
    }
    // stage ang rows (16 valid cols, pad to 32)
    for (int i = tid; i < 64 * 16; i += 128) {
        int r = i >> 4, cp = i & 15;
        int row = base + r; if (row >= A) row = A - 1;
        unsigned int u = 0;
        if (cp < 8) u = pack2bf(ang[row * 16 + cp * 2], ang[row * 16 + cp * 2 + 1]);
        ((unsigned int*)AngLds)[r * 16 + cp] = u;
    }
    __syncthreads();

    int wave = tid >> 5, lane = tid & 31;
    const unsigned short* Aw  = &Alds[wave * 16 * FD];
    const unsigned short* Anw = &AngLds[wave * 16 * 32];
    v8f acc[8], acc2[8];
#pragma unroll
    for (int t = 0; t < 8; ++t) { acc[t] = {}; acc2[t] = {}; }
#pragma unroll
    for (int kt = 0; kt < 4; ++kt) {
        v16bf af = load_a_frag(Aw, FD, kt * 32, lane);
#pragma unroll
        for (int t = 0; t < 8; ++t) {
            v16bf bfrag = load_b_packed(Wlds, t, kt, 4, lane);
            acc[t] = __builtin_amdgcn_wmma_f32_16x16x32_bf16(
                false, af, false, bfrag, (short)0, acc[t], false, false);
        }
    }
    {   // ang @ W_angle (K=16 padded to 32): one WMMA per n-tile
        v16bf af2 = load_a_frag(Anw, 32, 0, lane);
#pragma unroll
        for (int t = 0; t < 8; ++t) {
            v16bf bfrag = load_b_packed(WaLds, t, 0, 1, lane);
            acc2[t] = __builtin_amdgcn_wmma_f32_16x16x32_bf16(
                false, af2, false, bfrag, (short)0, acc2[t], false, false);
        }
    }

    int half = lane >> 4, nlo = lane & 15;
#pragma unroll
    for (int t = 0; t < 8; ++t) {
        int n = t * 16 + nlo;
        float bgn = bg[n];
#pragma unroll
        for (int r = 0; r < 8; ++r) {
            int lrow = wave * 16 + r + 8 * half;
            int row = base + lrow;
            if (row < A) {
                float g = sigmoidf_(acc[t][r] + bgn);
                float val = acc2[t][r] * g * fc_ik[row];
                atomicAdd(&agg[(size_t)jrow[lrow] * FD + n], val);
            }
        }
    }
}

// ---------------- generic 128x128 GEMM + swish (WMMA) ----------------
__global__ __launch_bounds__(128) void k_gemm_swish(
    const float* __restrict__ Arows, const unsigned short* __restrict__ WP,
    const float* __restrict__ bias, float* __restrict__ out, int M, int accumulate) {
    __shared__ __align__(32) unsigned short Wlds[FD * FD];
    __shared__ __align__(16) unsigned short Alds[64 * FD];
    int tid = threadIdx.x;
    int base = blockIdx.x * 64;

    for (int i = tid; i < 2048; i += 128) ((uint4*)Wlds)[i] = ((const uint4*)WP)[i];
    for (int i = tid; i < 64 * 16; i += 128) {
        int r = i >> 4, g = i & 15;
        int row = base + r; if (row >= M) row = M - 1;
        const float4* s4 = (const float4*)(Arows + (size_t)row * FD + g * 8);
        float4 x0 = s4[0], x1 = s4[1];
        uint4 u;
        u.x = pack2bf(x0.x, x0.y); u.y = pack2bf(x0.z, x0.w);
        u.z = pack2bf(x1.x, x1.y); u.w = pack2bf(x1.z, x1.w);
        *(uint4*)(Alds + r * FD + g * 8) = u;
    }
    __syncthreads();

    int wave = tid >> 5, lane = tid & 31;
    const unsigned short* Aw = &Alds[wave * 16 * FD];
    v8f acc[8];
#pragma unroll
    for (int t = 0; t < 8; ++t) acc[t] = {};
#pragma unroll
    for (int kt = 0; kt < 4; ++kt) {
        v16bf af = load_a_frag(Aw, FD, kt * 32, lane);
#pragma unroll
        for (int t = 0; t < 8; ++t) {
            v16bf bfrag = load_b_packed(Wlds, t, kt, 4, lane);
            acc[t] = __builtin_amdgcn_wmma_f32_16x16x32_bf16(
                false, af, false, bfrag, (short)0, acc[t], false, false);
        }
    }

    int half = lane >> 4, nlo = lane & 15;
#pragma unroll
    for (int t = 0; t < 8; ++t) {
        int n = t * 16 + nlo;
        float bn = bias[n];
#pragma unroll
        for (int r = 0; r < 8; ++r) {
            int row = base + wave * 16 + r + 8 * half;
            if (row < M) {
                float x = swishf_(acc[t][r] + bn);
                if (accumulate) out[(size_t)row * FD + n] += x;
                else            out[(size_t)row * FD + n] = x;
            }
        }
    }
}

// ------ concat([vi,vj,e]) @ W(384x128); out (+=/=) swish(.)*(ef0@W0) ---------
__global__ __launch_bounds__(128) void k_concat_gemm(
    const float* __restrict__ v, const float* __restrict__ e,
    const int* __restrict__ send, const int* __restrict__ recv,
    const unsigned short* __restrict__ W384P, const float* __restrict__ bias,
    const float* __restrict__ ef0, const unsigned short* __restrict__ W0P,
    float* __restrict__ out, int M, int accumulate) {
    __shared__ __align__(32) unsigned short Wlds[FD * FD];       // 32 KB (one K-chunk)
    __shared__ __align__(32) unsigned short W0Lds[8 * 32 * 16];  //  8 KB packed W0
    __shared__ __align__(16) unsigned short Alds[64 * FD];       // 16 KB
    __shared__ __align__(16) unsigned short Ef0Lds[64 * 32];     //  4 KB
    __shared__ int srow[64], rrow[64];
    int tid = threadIdx.x;
    int base = blockIdx.x * 64;
    if (tid < 64) {
        int row = base + tid; if (row >= M) row = M - 1;
        srow[tid] = send[row]; rrow[tid] = recv[row];
    }
    for (int i = tid; i < 512; i += 128) ((uint4*)W0Lds)[i] = ((const uint4*)W0P)[i];
    for (int i = tid; i < 64 * 16; i += 128) {
        int r = i >> 4, cp = i & 15;
        int row = base + r; if (row >= M) row = M - 1;
        int c0 = cp * 2;
        float a0 = (c0 < 5) ? ef0[(size_t)row * 5 + c0] : 0.f;
        float a1 = (c0 + 1 < 5) ? ef0[(size_t)row * 5 + c0 + 1] : 0.f;
        ((unsigned int*)Ef0Lds)[r * 16 + cp] = pack2bf(a0, a1);
    }

    int wave = tid >> 5, lane = tid & 31;
    v8f acc[8];
#pragma unroll
    for (int t = 0; t < 8; ++t) acc[t] = {};

    for (int c = 0; c < 3; ++c) {
        __syncthreads();
        for (int i = tid; i < 2048; i += 128)
            ((uint4*)Wlds)[i] = ((const uint4*)(W384P + (size_t)c * FD * FD))[i];
        for (int i = tid; i < 64 * 16; i += 128) {
            int r = i >> 4, g = i & 15;
            int row = base + r; if (row >= M) row = M - 1;
            const float* srcp = (c == 0) ? v + (size_t)srow[r] * FD
                              : (c == 1) ? v + (size_t)rrow[r] * FD
                                         : e + (size_t)row * FD;
            const float4* s4 = (const float4*)(srcp + g * 8);
            float4 x0 = s4[0], x1 = s4[1];
            uint4 u;
            u.x = pack2bf(x0.x, x0.y); u.y = pack2bf(x0.z, x0.w);
            u.z = pack2bf(x1.x, x1.y); u.w = pack2bf(x1.z, x1.w);
            *(uint4*)(Alds + r * FD + g * 8) = u;
        }
        __syncthreads();
        const unsigned short* Aw = &Alds[wave * 16 * FD];
#pragma unroll
        for (int kt = 0; kt < 4; ++kt) {
            v16bf af = load_a_frag(Aw, FD, kt * 32, lane);
#pragma unroll
            for (int t = 0; t < 8; ++t) {
                v16bf bfrag = load_b_packed(Wlds, t, kt, 4, lane);
                acc[t] = __builtin_amdgcn_wmma_f32_16x16x32_bf16(
                    false, af, false, bfrag, (short)0, acc[t], false, false);
            }
        }
    }

    // ef0 @ W0 via WMMA (K=5 padded to 32)
    v8f acc3[8];
#pragma unroll
    for (int t = 0; t < 8; ++t) acc3[t] = {};
    {
        const unsigned short* Ew = &Ef0Lds[wave * 16 * 32];
        v16bf af3 = load_a_frag(Ew, 32, 0, lane);
#pragma unroll
        for (int t = 0; t < 8; ++t) {
            v16bf bfrag = load_b_packed(W0Lds, t, 0, 1, lane);
            acc3[t] = __builtin_amdgcn_wmma_f32_16x16x32_bf16(
                false, af3, false, bfrag, (short)0, acc3[t], false, false);
        }
    }

    int half = lane >> 4, nlo = lane & 15;
#pragma unroll
    for (int t = 0; t < 8; ++t) {
        int n = t * 16 + nlo;
        float bn = bias[n];
#pragma unroll
        for (int r = 0; r < 8; ++r) {
            int row = base + wave * 16 + r + 8 * half;
            if (row < M) {
                float val = swishf_(acc[t][r] + bn) * acc3[t][r];
                if (accumulate) out[(size_t)row * FD + n] += val;
                else            out[(size_t)row * FD + n] = val;
            }
        }
    }
}

__global__ void k_scatter_v(const float* __restrict__ mbuf, const int* __restrict__ recv,
                            float* __restrict__ v, int E) {
    int i = blockIdx.x * 256 + threadIdx.x;
    if (i >= E * FD) return;
    int e = i >> 7, f = i & 127;
    atomicAdd(&v[(size_t)recv[e] * FD + f], mbuf[i]);
}

__global__ void k_final(const float* __restrict__ h, const float* __restrict__ W3,
                        const float* __restrict__ b3, float* __restrict__ out, int N) {
    int a = blockIdx.x * 256 + threadIdx.x;
    if (a >= N) return;
    float s = b3[0];
#pragma unroll 8
    for (int f = 0; f < FD; ++f) s += h[(size_t)a * FD + f] * W3[f];
    out[a] = s;
}

// ---------------- host launcher ----------------
extern "C" void kernel_launch(void* const* d_in, const int* in_sizes, int n_in,
                              void* d_out, int out_size, void* d_ws, size_t ws_size,
                              hipStream_t stream) {
    const int*   atomic_numbers = (const int*)  d_in[0];
    const int*   edge_index     = (const int*)  d_in[1];
    const float* edge_dist      = (const float*)d_in[2];
    const int*   tbi            = (const int*)  d_in[3];
    const float* norm_ik        = (const float*)d_in[4];
    const float* cos_angles     = (const float*)d_in[5];
    const int*   tnb            = (const int*)  d_in[6];
    const int*   tna            = (const int*)  d_in[7];
    const float* emb            = (const float*)d_in[8];
    const float* W_enc          = (const float*)d_in[9];
    const float* b_enc          = (const float*)d_in[10];
    const float* W_angle        = (const float*)d_in[11];
    const float* Wg             = (const float*)d_in[12];
    const float* bg             = (const float*)d_in[13];
    const float* W_tb           = (const float*)d_in[14];
    const float* b_tb           = (const float*)d_in[15];
    const float* We1            = (const float*)d_in[16];
    const float* be1            = (const float*)d_in[17];
    const float* We0            = (const float*)d_in[18];
    const float* Wa1            = (const float*)d_in[19];
    const float* ba1            = (const float*)d_in[20];
    const float* Wa0            = (const float*)d_in[21];
    const float* W1             = (const float*)d_in[22];
    const float* b1             = (const float*)d_in[23];
    const float* W2             = (const float*)d_in[24];
    const float* b2             = (const float*)d_in[25];
    const float* W3             = (const float*)d_in[26];
    const float* b3             = (const float*)d_in[27];

    const int N = in_sizes[0];
    const int E = in_sizes[2];
    const int A = in_sizes[4];
    const int G = in_sizes[6];
    const int NBk = 4;
    const size_t PK128 = 16384;   // packed ushorts per 128x128 chunk
    const size_t PK32  = 4096;    // packed ushorts per (K<=32)x128 chunk

    // ---- carve workspace ----
    char* wp = (char*)d_ws;
    auto carve = [&](size_t bytes) -> void* {
        void* p = (void*)wp;
        wp += (bytes + 255) & ~(size_t)255;
        return p;
    };
    float* ef0   = (float*)carve((size_t)E * 5 * 4);
    float* efeat = (float*)carve((size_t)E * FD * 4);
    float* vfeat = (float*)carve((size_t)N * FD * 4);
    float* ang   = (float*)carve((size_t)A * 16 * 4);
    float* fc_ik = (float*)carve((size_t)A * 4);
    int*   idx_ij = (int*)carve((size_t)A * 4);
    int*   idx_ik = (int*)carve((size_t)A * 4);
    float* agg   = (float*)carve((size_t)E * FD * 4);
    float* mbuf  = (float*)carve((size_t)E * FD * 4);
    float* h1    = (float*)carve((size_t)N * FD * 4);
    float* h2    = (float*)carve((size_t)N * FD * 4);
    int*   boff  = (int*)carve((size_t)G * 4);
    int*   acum  = (int*)carve((size_t)G * 4);
    unsigned short* WgP   = (unsigned short*)carve(NBk * PK128 * 2);
    unsigned short* WtbP  = (unsigned short*)carve(NBk * PK128 * 2);
    unsigned short* We1P  = (unsigned short*)carve(NBk * 3 * PK128 * 2);
    unsigned short* Wa1P  = (unsigned short*)carve(NBk * 3 * PK128 * 2);
    unsigned short* W1P   = (unsigned short*)carve(PK128 * 2);
    unsigned short* W2P   = (unsigned short*)carve(PK128 * 2);
    unsigned short* WangP = (unsigned short*)carve(NBk * PK32 * 2);
    unsigned short* We0P  = (unsigned short*)carve(NBk * PK32 * 2);
    unsigned short* Wa0P  = (unsigned short*)carve(NBk * PK32 * 2);
    (void)ws_size; (void)n_in; (void)out_size;

    const int* send = edge_index;
    const int* recv = edge_index + E;

    auto cdiv = [](int a, int b) { return (a + b - 1) / b; };

    // ---- weight pre-swizzle into WMMA fragment order ----
    k_pack_wB<<<cdiv(NBk * 1024, 256), 256, 0, stream>>>(Wg,   WgP,  NBk,     128, 4);
    k_pack_wB<<<cdiv(NBk * 1024, 256), 256, 0, stream>>>(W_tb, WtbP, NBk,     128, 4);
    k_pack_wB<<<cdiv(NBk * 3 * 1024, 256), 256, 0, stream>>>(We1, We1P, NBk * 3, 128, 4);
    k_pack_wB<<<cdiv(NBk * 3 * 1024, 256), 256, 0, stream>>>(Wa1, Wa1P, NBk * 3, 128, 4);
    k_pack_wB<<<cdiv(1024, 256), 256, 0, stream>>>(W1, W1P, 1, 128, 4);
    k_pack_wB<<<cdiv(1024, 256), 256, 0, stream>>>(W2, W2P, 1, 128, 4);
    k_pack_wB<<<cdiv(NBk * 256, 256), 256, 0, stream>>>(W_angle, WangP, NBk, 16, 1);
    k_pack_wB<<<cdiv(NBk * 256, 256), 256, 0, stream>>>(We0, We0P, NBk, 5, 1);
    k_pack_wB<<<cdiv(NBk * 256, 256), 256, 0, stream>>>(Wa0, Wa0P, NBk, 5, 1);

    // ---- one-time prep ----
    k_offsets<<<1, 64, 0, stream>>>(tnb, tna, G, boff, acum);
    k_tbi<<<cdiv(A, 256), 256, 0, stream>>>(tbi, boff, acum, G, A, idx_ij, idx_ik);
    k_rbf<<<cdiv(E, 256), 256, 0, stream>>>(edge_dist, ef0, E);
    k_edge_enc<<<cdiv(E * FD, 256), 256, 0, stream>>>(ef0, W_enc, b_enc, efeat, E);
    k_vinit<<<cdiv(N * FD, 256), 256, 0, stream>>>(atomic_numbers, emb, vfeat, N);
    k_angle<<<cdiv(A, 256), 256, 0, stream>>>(norm_ik, cos_angles, ang, fc_ik, A);

    // ---- message-passing blocks ----
    for (int b = 0; b < NBk; ++b) {
        hipMemsetAsync(agg, 0, (size_t)E * FD * 4, stream);

        k_gate_msg<<<cdiv(A, 64), 128, 0, stream>>>(
            vfeat, recv, idx_ik, idx_ij,
            WgP + (size_t)b * PK128, bg + b * FD,
            ang, WangP + (size_t)b * PK32, fc_ik, agg, A);

        k_gemm_swish<<<cdiv(E, 64), 128, 0, stream>>>(
            agg, WtbP + (size_t)b * PK128, b_tb + b * FD, efeat, E, 1);

        k_concat_gemm<<<cdiv(E, 64), 128, 0, stream>>>(
            vfeat, efeat, send, recv,
            We1P + (size_t)b * 3 * PK128, be1 + b * FD,
            ef0, We0P + (size_t)b * PK32, efeat, E, 1);

        k_concat_gemm<<<cdiv(E, 64), 128, 0, stream>>>(
            vfeat, efeat, send, recv,
            Wa1P + (size_t)b * 3 * PK128, ba1 + b * FD,
            ef0, Wa0P + (size_t)b * PK32, mbuf, E, 0);

        k_scatter_v<<<cdiv(E * FD, 256), 256, 0, stream>>>(mbuf, recv, vfeat, E);
    }

    // ---- readout MLP ----
    k_gemm_swish<<<cdiv(N, 64), 128, 0, stream>>>(vfeat, W1P, b1, h1, N, 0);
    k_gemm_swish<<<cdiv(N, 64), 128, 0, stream>>>(h1, W2P, b2, h2, N, 0);
    k_final<<<cdiv(N, 256), 256, 0, stream>>>(h2, W3, b3, (float*)d_out, N);
}
